// SocialLSTM_81535659147857
// MI455X (gfx1250) — matrix operand order repaired and checked by reading, not compiled, not run on hardware
//
#include <hip/hip_runtime.h>
#include <hip/hip_bf16.h>

typedef __bf16 bf16_t;
typedef __attribute__((ext_vector_type(16))) bf16_t v16bf;
typedef __attribute__((ext_vector_type(8)))  bf16_t v8bf;
typedef __attribute__((ext_vector_type(8)))  float  v8f;

#define APP   21
#define H     1024
#define E     512
#define SEQ   1024
#define INPD  44
#define K2    2048          // concat K  ([z|h0] or [h0n|h1])
#define N4    4096          // 4*H
#define NKT   64            // K2/32 k-tiles
#define NNT   256           // N4/16 n-tiles
#define SOC   (APP*H)       // 21504

#define NBLK  64
#define TPB   256
#define NTHREADS (NBLK*TPB) // 16384  -> 512 wave32 waves == 512 output tiles

#define PACK_ELEMS (NNT*NKT*32*16)   // 8,388,608 bf16 per layer

// ---- workspace byte offsets (all 256B aligned) ----
#define OFF_W0   ((size_t)0)
#define OFF_W1   (OFF_W0 + (size_t)PACK_ELEMS*2)
#define OFF_A0   (OFF_W1 + (size_t)PACK_ELEMS*2)
#define OFF_A1   (OFF_A0 + (size_t)32*K2*2)
#define OFF_G    (OFF_A1 + (size_t)32*K2*2)
#define OFF_H0   (OFF_G  + (size_t)32*N4*4)
#define OFF_C0   (OFF_H0 + (size_t)SOC*4)
#define OFF_H1   (OFF_C0 + (size_t)SOC*4)
#define OFF_C1   (OFF_H1 + (size_t)SOC*4)
#define OFF_BAR  (OFF_C1 + (size_t)SOC*4)

__device__ __forceinline__ float sigmoid_(float x) {
    return 1.0f / (1.0f + __expf(-x));
}

// Ticket grid barrier: counter is reset to 0 each launch by init kernel, each
// barrier adds exactly NBLK, so (ticket/NBLK+1)*NBLK is a consistent target.
__device__ __forceinline__ void grid_barrier(unsigned* cnt) {
    __threadfence();
    __syncthreads();
    if (threadIdx.x == 0) {
        unsigned ticket = __hip_atomic_fetch_add(cnt, 1u, __ATOMIC_ACQ_REL,
                                                 __HIP_MEMORY_SCOPE_AGENT);
        unsigned target = (ticket / NBLK + 1u) * NBLK;
        while (__hip_atomic_load(cnt, __ATOMIC_ACQUIRE,
                                 __HIP_MEMORY_SCOPE_AGENT) < target) {
            __builtin_amdgcn_s_sleep(1);
        }
    }
    __syncthreads();
    __threadfence();
}

// One 16x16 output tile per wave: gates[mt*16.., nt*16..] = act(32xK2) @ Wpack.
// A layout (16-bit A 16x32): lane<16 holds row (lane), K {0..7,16..23};
// lane>=16 holds row (lane-16), K {8..15,24..31}  -> two 16B loads.
// B prepacked in fragment order: 32B contiguous per lane per k-tile.
__device__ __forceinline__ void gemm_tiles(const bf16_t* __restrict__ act,
                                           const bf16_t* __restrict__ wpack,
                                           float* __restrict__ gates,
                                           int gwave, int lane) {
    const int mt    = gwave >> 8;      // 0..1
    const int nt    = gwave & 255;     // 0..255
    const int khalf = lane >> 4;       // 0 or 1
    const int row   = mt * 16 + (lane & 15);
    const bf16_t* arow  = act + (size_t)row * K2 + khalf * 8;
    const bf16_t* bbase = wpack + (size_t)nt * NKT * 512 + (size_t)lane * 16;

    v8f acc = {0.f, 0.f, 0.f, 0.f, 0.f, 0.f, 0.f, 0.f};
    union Frag { v16bf v; v8bf h[2]; };

    for (int kt = 0; kt < NKT; ++kt) {
        Frag A, B;
        A.h[0] = *(const v8bf*)(arow + kt * 32);        // K = kt*32+khalf*8+0..7
        A.h[1] = *(const v8bf*)(arow + kt * 32 + 16);   // K = +16
        const bf16_t* bp = bbase + (size_t)kt * 512;
        B.h[0] = *(const v8bf*)(bp);
        B.h[1] = *(const v8bf*)(bp + 8);
        acc = __builtin_amdgcn_wmma_f32_16x16x32_bf16(
                  false, A.v, false, B.v, (short)0, acc, false, false);
    }
    // D layout: vgpr v -> row mt*16 + (lane>=16 ? 8 : 0) + v, col nt*16+(lane&15)
    const int rbase = mt * 16 + khalf * 8;
    const int col   = nt * 16 + (lane & 15);
#pragma unroll
    for (int v = 0; v < 8; ++v)
        gates[(size_t)(rbase + v) * N4 + col] = acc[v];
}

// Pack [W_ih | W_hh] (each 4096x1024 fp32 row-major) into bf16 B-fragment
// order:  r = ((nt*NKT + kt)*32 + lane)*16 + e ;  B[k_local, col] with
// col = lane&15, k_local = (lane>>4)*8 + (e&7) + (e>>3)*16, n = nt*16+col, k = kt*32+k_local.
__global__ void prepack_kernel(const float* __restrict__ Wih0,
                               const float* __restrict__ Whh0,
                               const float* __restrict__ Wih1,
                               const float* __restrict__ Whh1,
                               bf16_t* __restrict__ w0,
                               bf16_t* __restrict__ w1) {
    const long total = 2L * PACK_ELEMS;
    for (long idx = (long)blockIdx.x * blockDim.x + threadIdx.x; idx < total;
         idx += (long)gridDim.x * blockDim.x) {
        int layer = (int)(idx >> 23);
        int r     = (int)(idx & (PACK_ELEMS - 1));
        int e     = r & 15;
        int lane  = (r >> 4) & 31;
        int kt    = (r >> 9) & 63;
        int nt    = r >> 15;
        int col   = lane & 15;
        int n     = nt * 16 + col;
        int k     = kt * 32 + (lane >> 4) * 8 + (e & 7) + (e >> 3) * 16;
        const float* Wi = layer ? Wih1 : Wih0;
        const float* Wh = layer ? Whh1 : Whh0;
        float v = (k < H) ? Wi[(size_t)n * H + k] : Wh[(size_t)n * H + (k - H)];
        (layer ? w1 : w0)[r] = (bf16_t)v;
    }
}

__global__ void init_kernel(float* h0, float* c0, float* h1, float* c1,
                            bf16_t* a0, bf16_t* a1, unsigned* bar) {
    int tid = blockIdx.x * blockDim.x + threadIdx.x;
    int nt  = gridDim.x * blockDim.x;
    for (int i = tid; i < SOC; i += nt) { h0[i] = 0.f; c0[i] = 0.f; h1[i] = 0.f; c1[i] = 0.f; }
    for (int i = tid; i < 32 * K2; i += nt) { a0[i] = (bf16_t)0.f; a1[i] = (bf16_t)0.f; }
    if (tid < 64) bar[tid] = 0u;
}

__global__ void __launch_bounds__(TPB, 1)
lstm_persistent(const float* __restrict__ x,
                const float* __restrict__ W_ie, const float* __restrict__ b_ie,
                const float* __restrict__ W_te, const float* __restrict__ b_te,
                const float* __restrict__ b_ih0, const float* __restrict__ b_hh0,
                const float* __restrict__ b_ih1, const float* __restrict__ b_hh1,
                const float* __restrict__ W_out, const float* __restrict__ b_out,
                const bf16_t* __restrict__ w0pack, const bf16_t* __restrict__ w1pack,
                bf16_t* __restrict__ act0, bf16_t* __restrict__ act1,
                float* __restrict__ gates,
                float* __restrict__ h0, float* __restrict__ c0,
                float* __restrict__ h1, float* __restrict__ c1,
                unsigned* __restrict__ bar, float* __restrict__ out) {
    const int tid   = threadIdx.x;
    const int gtid  = blockIdx.x * TPB + tid;
    const int gwave = gtid >> 5;      // 0..511
    const int lane  = tid & 31;
    __shared__ float red[TPB];

    for (int t = 0; t < SEQ; ++t) {
        // ---------- Phase A: out[t-1], te, ie ----------
        if (t > 0 && blockIdx.x == 0) {
            float s = 0.f;
            for (int k = tid; k < SOC; k += TPB) s = fmaf(W_out[k], h1[k], s);
            red[tid] = s; __syncthreads();
            for (int w = TPB / 2; w > 0; w >>= 1) {
                if (tid < w) red[tid] += red[tid + w];
                __syncthreads();
            }
            if (tid == 0) out[t - 1] = red[0] + b_out[0];
        }
        {   // te[e] : one element per wave, fp32 GEMV over social = h1 flat
            const int e = gwave;
            const float* wrow = W_te + (size_t)e * SOC;
            float s = 0.f;
            for (int k = lane; k < SOC; k += 32) s = fmaf(wrow[k], h1[k], s);
#pragma unroll
            for (int off = 16; off > 0; off >>= 1) s += __shfl_xor(s, off, 32);
            s = fmaxf(s + b_te[e], 0.f);
            if (lane < APP) act0[(size_t)lane * K2 + E + e] = (bf16_t)s;
        }
        {   // ie[m][e] : xt = [x[t][0:43], x[t][43+m]]
            const float* xt = x + (size_t)t * 64;
            for (int idx = gtid; idx < APP * E; idx += NTHREADS) {
                int m = idx >> 9, e = idx & (E - 1);
                const float* wrow = W_ie + (size_t)e * INPD;
                float s = b_ie[e];
                for (int j = 0; j < 43; ++j) s = fmaf(wrow[j], xt[j], s);
                s = fmaf(wrow[43], xt[43 + m], s);
                act0[(size_t)m * K2 + e] = (bf16_t)fmaxf(s, 0.f);
            }
        }
        grid_barrier(bar);
        // ---------- Phase B: gates0 = [z|h0] @ W0^T  (WMMA bf16) ----------
        gemm_tiles(act0, w0pack, gates, gwave, lane);
        grid_barrier(bar);
        // ---------- Phase C: LSTM cell 0 ----------
        for (int idx = gtid; idx < SOC; idx += NTHREADS) {
            int m = idx >> 10, j = idx & (H - 1);
            const float* g = gates + (size_t)m * N4;
            float gi = g[j]         + b_ih0[j]         + b_hh0[j];
            float gf = g[H + j]     + b_ih0[H + j]     + b_hh0[H + j];
            float gg = g[2 * H + j] + b_ih0[2 * H + j] + b_hh0[2 * H + j];
            float go = g[3 * H + j] + b_ih0[3 * H + j] + b_hh0[3 * H + j];
            float cn = sigmoid_(gf) * c0[idx] + sigmoid_(gi) * tanhf(gg);
            float hn = sigmoid_(go) * tanhf(cn);
            c0[idx] = cn; h0[idx] = hn;
            bf16_t hb = (bf16_t)hn;
            act1[(size_t)m * K2 + j]     = hb;   // layer-1 input this step
            act0[(size_t)m * K2 + H + j] = hb;   // h0 part for next step
        }
        grid_barrier(bar);
        // ---------- Phase D: gates1 = [h0n|h1] @ W1^T  (WMMA bf16) ----------
        gemm_tiles(act1, w1pack, gates, gwave, lane);
        grid_barrier(bar);
        // ---------- Phase E: LSTM cell 1 ----------
        for (int idx = gtid; idx < SOC; idx += NTHREADS) {
            int m = idx >> 10, j = idx & (H - 1);
            const float* g = gates + (size_t)m * N4;
            float gi = g[j]         + b_ih1[j]         + b_hh1[j];
            float gf = g[H + j]     + b_ih1[H + j]     + b_hh1[H + j];
            float gg = g[2 * H + j] + b_ih1[2 * H + j] + b_hh1[2 * H + j];
            float go = g[3 * H + j] + b_ih1[3 * H + j] + b_hh1[3 * H + j];
            float cn = sigmoid_(gf) * c1[idx] + sigmoid_(gi) * tanhf(gg);
            float hn = sigmoid_(go) * tanhf(cn);
            c1[idx] = cn; h1[idx] = hn;                      // social for next step
            act1[(size_t)m * K2 + H + j] = (bf16_t)hn;       // h1 part for next step
        }
        grid_barrier(bar);
    }
    // ---------- Final: out[SEQ-1] + state dump ----------
    if (blockIdx.x == 0) {
        float s = 0.f;
        for (int k = tid; k < SOC; k += TPB) s = fmaf(W_out[k], h1[k], s);
        red[tid] = s; __syncthreads();
        for (int w = TPB / 2; w > 0; w >>= 1) {
            if (tid < w) red[tid] += red[tid + w];
            __syncthreads();
        }
        if (tid == 0) out[SEQ - 1] = red[0] + b_out[0];
    }
    for (int idx = gtid; idx < SOC; idx += NTHREADS) {
        out[SEQ + idx]           = h0[idx];   // hidden[0]
        out[SEQ + SOC + idx]     = h1[idx];   // hidden[1]
        out[SEQ + 2 * SOC + idx] = c0[idx];   // cell[0]
        out[SEQ + 3 * SOC + idx] = c1[idx];   // cell[1]
    }
}

extern "C" void kernel_launch(void* const* d_in, const int* in_sizes, int n_in,
                              void* d_out, int out_size, void* d_ws, size_t ws_size,
                              hipStream_t stream) {
    (void)in_sizes; (void)n_in; (void)out_size; (void)ws_size;
    const float* x     = (const float*)d_in[0];
    const float* W_ie  = (const float*)d_in[1];
    const float* b_ie  = (const float*)d_in[2];
    const float* W_te  = (const float*)d_in[3];
    const float* b_te  = (const float*)d_in[4];
    const float* W_ih0 = (const float*)d_in[5];
    const float* W_hh0 = (const float*)d_in[6];
    const float* b_ih0 = (const float*)d_in[7];
    const float* b_hh0 = (const float*)d_in[8];
    const float* W_ih1 = (const float*)d_in[9];
    const float* W_hh1 = (const float*)d_in[10];
    const float* b_ih1 = (const float*)d_in[11];
    const float* b_hh1 = (const float*)d_in[12];
    const float* W_out = (const float*)d_in[13];
    const float* b_out = (const float*)d_in[14];

    char* ws = (char*)d_ws;
    bf16_t*   w0pack = (bf16_t*)(ws + OFF_W0);
    bf16_t*   w1pack = (bf16_t*)(ws + OFF_W1);
    bf16_t*   act0   = (bf16_t*)(ws + OFF_A0);
    bf16_t*   act1   = (bf16_t*)(ws + OFF_A1);
    float*    gates  = (float*)  (ws + OFF_G);
    float*    h0     = (float*)  (ws + OFF_H0);
    float*    c0     = (float*)  (ws + OFF_C0);
    float*    h1     = (float*)  (ws + OFF_H1);
    float*    c1     = (float*)  (ws + OFF_C1);
    unsigned* bar    = (unsigned*)(ws + OFF_BAR);
    float*    out    = (float*)d_out;

    init_kernel<<<128, 256, 0, stream>>>(h0, c0, h1, c1, act0, act1, bar);
    prepack_kernel<<<1024, 256, 0, stream>>>(W_ih0, W_hh0, W_ih1, W_hh1, w0pack, w1pack);
    lstm_persistent<<<NBLK, TPB, 0, stream>>>(
        x, W_ie, b_ie, W_te, b_te, b_ih0, b_hh0, b_ih1, b_hh1, W_out, b_out,
        w0pack, w1pack, act0, act1, gates, h0, c0, h1, c1, bar, out);
}
